// EGNNDynamicVars_87454124081344
// MI455X (gfx1250) — matrix operand order repaired and checked by reading, not compiled, not run on hardware
//
#include <hip/hip_runtime.h>

typedef __attribute__((ext_vector_type(16))) _Float16 v16h;
typedef __attribute__((ext_vector_type(8)))  _Float16 v8h;
typedef __attribute__((ext_vector_type(8)))  float    v8f;
typedef __attribute__((ext_vector_type(4)))  float    v4f;

#define HID 64
#define NN  512

__device__ __forceinline__ float silu_f(float x) {
  return x / (1.0f + __expf(-x));
}

__device__ __forceinline__ v8f wmma16(v16h a, v16h b, v8f c) {
  // D = A(16x32 f16) * B(32x16 f16) + C(16x16 f32)
  return __builtin_amdgcn_wmma_f32_16x16x32_f16(false, a, false, b, (short)0, c,
                                                false, false);
}

// A-fragment: lane L holds M = L%16, K-halves {kbase.. kbase+7, kbase+16..kbase+23}
// caller passes pointer to element [M][kbase]; second 8 halves live at +16.
__device__ __forceinline__ v16h load_afrag(const _Float16* p) {
  v8h lo = *(const v8h*)p;
  v8h hi = *(const v8h*)(p + 16);
  v16h a;
#pragma unroll
  for (int i = 0; i < 8; ++i) { a[i] = lo[i]; a[i + 8] = hi[i]; }
  return a;
}

// B-fragment pre-swizzled: 16 contiguous halves per lane.
__device__ __forceinline__ v16h load_bfrag(const _Float16* p) {
  v8h lo = *(const v8h*)p;
  v8h hi = *(const v8h*)(p + 8);
  v16h b;
#pragma unroll
  for (int i = 0; i < 8; ++i) { b[i] = lo[i]; b[i + 8] = hi[i]; }
  return b;
}

// Convert W[K][64] (f32, row-major) into per-lane B-fragment layout:
// dst[((kc*4+nt)*32+lane)*16 + i] = W[kc*32 + (lane&16?8:0) + (i&8?16:0) + (i&7)][nt*16 + lane&15]
__device__ static void swizzleB(const float* __restrict__ W, _Float16* dst,
                                int nkc, int tid, int nthreads) {
  const int total = nkc * 4 * 32 * 16;
  for (int idx = tid; idx < total; idx += nthreads) {
    const int i    = idx & 15;
    const int lane = (idx >> 4) & 31;
    const int nt   = (idx >> 9) & 3;
    const int kc   = idx >> 11;
    const int k = kc * 32 + ((lane & 16) ? 8 : 0) + ((i & 8) ? 16 : 0) + (i & 7);
    const int n = nt * 16 + (lane & 15);
    dst[idx] = (_Float16)W[k * HID + n];
  }
}

// ---------------------------------------------------------------------------
// Edge kernel: fused per-edge MLP chain, 16 edges per wave per tile.
// ---------------------------------------------------------------------------
__global__ __launch_bounds__(128) void egnn_edge_kernel(
    const _Float16* __restrict__ h16, const float* __restrict__ coordW,
    const int* __restrict__ send, const int* __restrict__ recv,
    const float* __restrict__ eW1, const float* __restrict__ eb1,
    const float* __restrict__ eW2, const float* __restrict__ eb2,
    const float* __restrict__ cW1, const float* __restrict__ cb1,
    const float* __restrict__ cW2,
    float* __restrict__ agg, float* __restrict__ aggv, int E) {
  __shared__ _Float16 wB1[4 * 4 * 32 * 16];   // eW1 (K=128)
  __shared__ _Float16 wB2[2 * 4 * 32 * 16];   // eW2 (K=64)
  __shared__ _Float16 wBc[2 * 4 * 32 * 16];   // cW1 (K=64)
  __shared__ float w1r_s[HID], eb1_s[HID], eb2_s[HID], cb1_s[HID], cw2_s[HID];
  __shared__ _Float16 mtile[4][16][HID];      // per-wave layout-conversion tile
  __shared__ float radS[4][16];
  __shared__ float diffS[4][16][3];
  __shared__ int   colS[4][16];
  __shared__ float cmS[4][16];

  const int tid = threadIdx.x;
  swizzleB(eW1, wB1, 4, tid, blockDim.x);
  swizzleB(eW2, wB2, 2, tid, blockDim.x);
  swizzleB(cW1, wBc, 2, tid, blockDim.x);
  for (int i = tid; i < HID; i += blockDim.x) {
    w1r_s[i] = eW1[128 * HID + i];   // radial row of eW1
    eb1_s[i] = eb1[i];
    eb2_s[i] = eb2[i];
    cb1_s[i] = cb1[i];
    cw2_s[i] = cW2[i];
  }
  __syncthreads();

  const int lane = tid & 31;
  const int w    = tid >> 5;
  const int half = lane >> 4;
  const int n0   = lane & 15;
  const int ab   = half * 8;       // A-fragment K base within a 32-chunk
  const int Mme  = lane & 15;

  const int ntiles = (E + 15) >> 4;
  const int gw = blockIdx.x * (blockDim.x >> 5) + w;
  const int nw = gridDim.x * (blockDim.x >> 5);

  for (int tile = gw; tile < ntiles; tile += nw) {
    const int  e  = tile * 16 + Mme;
    const bool vE = (e < E);
    const int  ec = vE ? e : (E - 1);
    const int nodeR = send[ec];
    const int nodeC = recv[ec];

    if (lane < 16) {
      float dx = coordW[nodeR * 3 + 0] - coordW[nodeC * 3 + 0];
      float dy = coordW[nodeR * 3 + 1] - coordW[nodeC * 3 + 1];
      float dz = coordW[nodeR * 3 + 2] - coordW[nodeC * 3 + 2];
      diffS[w][lane][0] = dx; diffS[w][lane][1] = dy; diffS[w][lane][2] = dz;
      radS[w][lane] = dx * dx + dy * dy + dz * dz;
      colS[w][lane] = nodeC;
    }

    // GEMM1: m = silu(ein @ eW1 + eb1). radial column (K=128) folded into C-init.
    v8f acc[4];
#pragma unroll
    for (int nt = 0; nt < 4; ++nt)
#pragma unroll
      for (int r = 0; r < 8; ++r) {
        const int M = r + 8 * half;
        const int n = nt * 16 + n0;
        acc[nt][r] = radS[w][M] * w1r_s[n] + eb1_s[n];
      }
#pragma unroll
    for (int kc = 0; kc < 4; ++kc) {
      const int node = (kc < 2) ? nodeR : nodeC;      // [h(row), h(col)] concat
      v16h a = load_afrag(h16 + node * HID + (kc & 1) * 32 + ab);
#pragma unroll
      for (int nt = 0; nt < 4; ++nt) {
        v16h b = load_bfrag(&wB1[((kc * 4 + nt) * 32 + lane) * 16]);
        acc[nt] = wmma16(a, b, acc[nt]);
      }
    }
#pragma unroll
    for (int nt = 0; nt < 4; ++nt)
#pragma unroll
      for (int r = 0; r < 8; ++r)
        mtile[w][r + 8 * half][nt * 16 + n0] = (_Float16)silu_f(acc[nt][r]);

    // GEMM2: ef = silu(m @ eW2 + eb2)
    v8f acc2[4];
#pragma unroll
    for (int nt = 0; nt < 4; ++nt)
#pragma unroll
      for (int r = 0; r < 8; ++r) acc2[nt][r] = eb2_s[nt * 16 + n0];
#pragma unroll
    for (int kc = 0; kc < 2; ++kc) {
      v16h a = load_afrag(&mtile[w][Mme][kc * 32 + ab]);
#pragma unroll
      for (int nt = 0; nt < 4; ++nt) {
        v16h b = load_bfrag(&wB2[((kc * 4 + nt) * 32 + lane) * 16]);
        acc2[nt] = wmma16(a, b, acc2[nt]);
      }
    }
    // ef -> mtile (overwrite, m is dead) + scatter-sum into agg[col]
#pragma unroll
    for (int r = 0; r < 8; ++r) {
      const int  M     = r + 8 * half;
      const int  cdest = colS[w][M];
      const bool vM    = (tile * 16 + M) < E;
#pragma unroll
      for (int nt = 0; nt < 4; ++nt) {
        float ef = silu_f(acc2[nt][r]);
        mtile[w][M][nt * 16 + n0] = (_Float16)ef;
        if (vM) atomicAdd(&agg[cdest * HID + nt * 16 + n0], ef);
      }
    }

    // GEMM3 + cm = silu(ef @ cW1 + cb1) @ cW2   (per-edge scalar)
    v8f acc3[4];
#pragma unroll
    for (int nt = 0; nt < 4; ++nt)
#pragma unroll
      for (int r = 0; r < 8; ++r) acc3[nt][r] = cb1_s[nt * 16 + n0];
#pragma unroll
    for (int kc = 0; kc < 2; ++kc) {
      v16h a = load_afrag(&mtile[w][Mme][kc * 32 + ab]);
#pragma unroll
      for (int nt = 0; nt < 4; ++nt) {
        v16h b = load_bfrag(&wBc[((kc * 4 + nt) * 32 + lane) * 16]);
        acc3[nt] = wmma16(a, b, acc3[nt]);
      }
    }
#pragma unroll
    for (int r = 0; r < 8; ++r) {
      float part = 0.f;
#pragma unroll
      for (int nt = 0; nt < 4; ++nt)
        part += silu_f(acc3[nt][r]) * cw2_s[nt * 16 + n0];
      part += __shfl_xor(part, 1, 32);
      part += __shfl_xor(part, 2, 32);
      part += __shfl_xor(part, 4, 32);
      part += __shfl_xor(part, 8, 32);
      if (n0 == 0) cmS[w][r + 8 * half] = part;
    }
    if (lane < 16 && vE) {
      const float cm = cmS[w][lane];
#pragma unroll
      for (int d = 0; d < 3; ++d) {
        float t = diffS[w][lane][d] * cm;
        t = fminf(fmaxf(t, -100.f), 100.f);
        atomicAdd(&aggv[nodeC * 3 + d], t);
      }
    }
  }
}

// ---------------------------------------------------------------------------
// Node kernel: node MLP (residual), velocity scale, coord/vel update.
// ---------------------------------------------------------------------------
__global__ __launch_bounds__(128) void egnn_node_kernel(
    const float* __restrict__ h_in, const _Float16* __restrict__ h16_in,
    const float* __restrict__ agg, const float* __restrict__ aggv,
    const float* __restrict__ cnt,
    float* __restrict__ coordW, float* __restrict__ velW,
    const float* __restrict__ nW1, const float* __restrict__ nb1,
    const float* __restrict__ nW2, const float* __restrict__ nb2,
    const float* __restrict__ vW1, const float* __restrict__ vb1,
    const float* __restrict__ vW2, const float* __restrict__ vb2,
    float* __restrict__ h_out, _Float16* __restrict__ h16_out) {
  __shared__ _Float16 wN1[4 * 4 * 32 * 16];
  __shared__ _Float16 wN2[2 * 4 * 32 * 16];
  __shared__ _Float16 wV1[2 * 4 * 32 * 16];
  __shared__ float nb1_s[HID], nb2_s[HID], vb1_s[HID], vw2_s[HID];
  __shared__ _Float16 mtile[4][16][HID];
  __shared__ float vsS[4][16];

  const int tid = threadIdx.x;
  swizzleB(nW1, wN1, 4, tid, blockDim.x);
  swizzleB(nW2, wN2, 2, tid, blockDim.x);
  swizzleB(vW1, wV1, 2, tid, blockDim.x);
  for (int i = tid; i < HID; i += blockDim.x) {
    nb1_s[i] = nb1[i]; nb2_s[i] = nb2[i]; vb1_s[i] = vb1[i]; vw2_s[i] = vW2[i];
  }
  __syncthreads();

  const int lane = tid & 31;
  const int w    = tid >> 5;
  const int half = lane >> 4;
  const int n0   = lane & 15;
  const int ab   = half * 8;
  const int Mme  = lane & 15;
  const int ntiles = NN / 16;
  const int gw = blockIdx.x * (blockDim.x >> 5) + w;
  const int nw = gridDim.x * (blockDim.x >> 5);
  const float vb2v = vb2[0];

  for (int tile = gw; tile < ntiles; tile += nw) {
    const int nodeA = tile * 16 + Mme;

    // node_in = [h, agg] (K=128); kc 0,1 from f16 h; kc 2,3 convert f32 agg
    v8f acc[4];
#pragma unroll
    for (int nt = 0; nt < 4; ++nt)
#pragma unroll
      for (int r = 0; r < 8; ++r) acc[nt][r] = nb1_s[nt * 16 + n0];
#pragma unroll
    for (int kc = 0; kc < 4; ++kc) {
      v16h a;
      if (kc < 2) {
        a = load_afrag(h16_in + nodeA * HID + kc * 32 + ab);
      } else {
        const float* ap = agg + nodeA * HID + (kc - 2) * 32 + ab;
        v4f f0 = *(const v4f*)(ap);
        v4f f1 = *(const v4f*)(ap + 4);
        v4f f2 = *(const v4f*)(ap + 16);
        v4f f3 = *(const v4f*)(ap + 20);
#pragma unroll
        for (int i = 0; i < 4; ++i) {
          a[i]      = (_Float16)f0[i];
          a[i + 4]  = (_Float16)f1[i];
          a[i + 8]  = (_Float16)f2[i];
          a[i + 12] = (_Float16)f3[i];
        }
      }
#pragma unroll
      for (int nt = 0; nt < 4; ++nt) {
        v16h b = load_bfrag(&wN1[((kc * 4 + nt) * 32 + lane) * 16]);
        acc[nt] = wmma16(a, b, acc[nt]);
      }
    }
#pragma unroll
    for (int nt = 0; nt < 4; ++nt)
#pragma unroll
      for (int r = 0; r < 8; ++r)
        mtile[w][r + 8 * half][nt * 16 + n0] = (_Float16)silu_f(acc[nt][r]);

    v8f acc2[4];
#pragma unroll
    for (int nt = 0; nt < 4; ++nt)
#pragma unroll
      for (int r = 0; r < 8; ++r) acc2[nt][r] = nb2_s[nt * 16 + n0];
#pragma unroll
    for (int kc = 0; kc < 2; ++kc) {
      v16h a = load_afrag(&mtile[w][Mme][kc * 32 + ab]);
#pragma unroll
      for (int nt = 0; nt < 4; ++nt) {
        v16h b = load_bfrag(&wN2[((kc * 4 + nt) * 32 + lane) * 16]);
        acc2[nt] = wmma16(a, b, acc2[nt]);
      }
    }
    // residual h update
#pragma unroll
    for (int r = 0; r < 8; ++r) {
      const int M  = r + 8 * half;
      const int nd = tile * 16 + M;
#pragma unroll
      for (int nt = 0; nt < 4; ++nt) {
        const int n = nt * 16 + n0;
        float v = acc2[nt][r] + h_in[nd * HID + n];
        h_out[nd * HID + n]   = v;
        h16_out[nd * HID + n] = (_Float16)v;
      }
    }

    // vel_scale = silu(h @ vW1 + vb1) @ vW2 + vb2
    v8f acc3[4];
#pragma unroll
    for (int nt = 0; nt < 4; ++nt)
#pragma unroll
      for (int r = 0; r < 8; ++r) acc3[nt][r] = vb1_s[nt * 16 + n0];
#pragma unroll
    for (int kc = 0; kc < 2; ++kc) {
      v16h a = load_afrag(h16_in + nodeA * HID + kc * 32 + ab);
#pragma unroll
      for (int nt = 0; nt < 4; ++nt) {
        v16h b = load_bfrag(&wV1[((kc * 4 + nt) * 32 + lane) * 16]);
        acc3[nt] = wmma16(a, b, acc3[nt]);
      }
    }
#pragma unroll
    for (int r = 0; r < 8; ++r) {
      float part = 0.f;
#pragma unroll
      for (int nt = 0; nt < 4; ++nt)
        part += silu_f(acc3[nt][r]) * vw2_s[nt * 16 + n0];
      part += __shfl_xor(part, 1, 32);
      part += __shfl_xor(part, 2, 32);
      part += __shfl_xor(part, 4, 32);
      part += __shfl_xor(part, 8, 32);
      if (n0 == 0) vsS[w][r + 8 * half] = part + vb2v;
    }
    if (lane < 16) {
      const int nd = tile * 16 + lane;
      const float vs    = vsS[w][lane];
      const float denom = fmaxf(cnt[nd], 1.0f);
#pragma unroll
      for (int d = 0; d < 3; ++d) {
        float nv = aggv[nd * 3 + d] / denom + vs * velW[nd * 3 + d];
        float nc = coordW[nd * 3 + d] + nv;
        velW[nd * 3 + d]   = nv;
        coordW[nd * 3 + d] = nc;
      }
    }
  }
}

// ---------------------------------------------------------------------------
// Utility kernels
// ---------------------------------------------------------------------------
__global__ void fill_zero_k(float* p, int n) {
  int i = blockIdx.x * blockDim.x + threadIdx.x;
  if (i < n) p[i] = 0.f;
}

__global__ void count_recv_k(const int* __restrict__ recv, float* cnt, int E) {
  int i = blockIdx.x * blockDim.x + threadIdx.x;
  if (i < E) atomicAdd(&cnt[recv[i]], 1.0f);
}

__global__ void embed_init_k(const float* __restrict__ hin,
                             const float* __restrict__ embW,
                             const float* __restrict__ embb,
                             const float* __restrict__ coord_in,
                             const float* __restrict__ vel_in,
                             float* h, _Float16* h16, float* coordW, float* velW) {
  int i = blockIdx.x * blockDim.x + threadIdx.x;
  if (i < NN * HID) {
    int node = i >> 6, n = i & 63;
    float v = hin[node] * embW[n] + embb[n];
    h[i] = v;
    h16[i] = (_Float16)v;
  }
  if (i < NN * 3) {
    coordW[i] = coord_in[i];
    velW[i]   = vel_in[i];
  }
}

__global__ void copy_out_k(const float* __restrict__ h,
                           const float* __restrict__ coordW,
                           const float* __restrict__ velW,
                           float* out, int out_size) {
  int i = blockIdx.x * blockDim.x + threadIdx.x;
  if (i >= out_size) return;
  if (i < NN * HID)                out[i] = h[i];
  else if (i < NN * HID + NN * 3)  out[i] = coordW[i - NN * HID];
  else if (i < NN * HID + 2*NN*3)  out[i] = velW[i - NN * HID - NN * 3];
}

// ---------------------------------------------------------------------------
// Host launcher
// ---------------------------------------------------------------------------
extern "C" void kernel_launch(void* const* d_in, const int* in_sizes, int n_in,
                              void* d_out, int out_size, void* d_ws, size_t ws_size,
                              hipStream_t stream) {
  if (n_in < 7 + 4 * 15) return;

  const float* hin      = (const float*)d_in[0];
  const float* coord_in = (const float*)d_in[1];
  const float* vel_in   = (const float*)d_in[2];
  const int*   send     = (const int*)d_in[3];
  const int*   recv     = (const int*)d_in[4];
  const float* embW     = (const float*)d_in[5];
  const float* embb     = (const float*)d_in[6];
  const int E = in_sizes[3];

  char* ws = (char*)d_ws;
  size_t off = 0;
  auto take = [&](size_t bytes) -> char* {
    char* p = ws + off;
    off = (off + bytes + 255) & ~(size_t)255;
    return p;
  };
  float*    hA     = (float*)take(NN * HID * 4);
  float*    hB     = (float*)take(NN * HID * 4);
  _Float16* h16A   = (_Float16*)take(NN * HID * 2);
  _Float16* h16B   = (_Float16*)take(NN * HID * 2);
  float*    coordW = (float*)take(NN * 3 * 4);
  float*    velW   = (float*)take(NN * 3 * 4);
  float*    agg    = (float*)take(NN * HID * 4);
  float*    aggv   = (float*)take(NN * 3 * 4);
  float*    cnt    = (float*)take(NN * 4);
  (void)ws_size;

  fill_zero_k<<<(NN + 255) / 256, 256, 0, stream>>>(cnt, NN);
  count_recv_k<<<(E + 255) / 256, 256, 0, stream>>>(recv, cnt, E);
  embed_init_k<<<(NN * HID + 255) / 256, 256, 0, stream>>>(
      hin, embW, embb, coord_in, vel_in, hA, h16A, coordW, velW);

  float*    hc   = hA;  _Float16* h16c = h16A;
  float*    hn   = hB;  _Float16* h16n = h16B;

  for (int l = 0; l < 4; ++l) {
    const float* const* P = (const float* const*)(d_in + 7 + 15 * l);
    // pytree (sorted-key) order within a layer dict:
    const float *cW1p = P[0],  *cW2p = P[1],  *cb1p = P[2];
    const float *eW1p = P[3],  *eW2p = P[4],  *eb1p = P[5],  *eb2p = P[6];
    const float *nW1p = P[7],  *nW2p = P[8],  *nb1p = P[9],  *nb2p = P[10];
    const float *vW1p = P[11], *vW2p = P[12], *vb1p = P[13], *vb2p = P[14];

    fill_zero_k<<<(NN * HID + 255) / 256, 256, 0, stream>>>(agg, NN * HID);
    fill_zero_k<<<(NN * 3 + 255) / 256, 256, 0, stream>>>(aggv, NN * 3);

    egnn_edge_kernel<<<768, 128, 0, stream>>>(
        h16c, coordW, send, recv, eW1p, eb1p, eW2p, eb2p, cW1p, cb1p, cW2p,
        agg, aggv, E);

    egnn_node_kernel<<<8, 128, 0, stream>>>(
        hc, h16c, agg, aggv, cnt, coordW, velW,
        nW1p, nb1p, nW2p, nb2p, vW1p, vb1p, vW2p, vb2p, hn, h16n);

    float* tf = hc; hc = hn; hn = tf;
    _Float16* th = h16c; h16c = h16n; h16n = th;
  }

  copy_out_k<<<(out_size + 255) / 256, 256, 0, stream>>>(
      hc, coordW, velW, (float*)d_out, out_size);
}
